// MambaBlock_83425444757957
// MI455X (gfx1250) — matrix operand order repaired and checked by reading, not compiled, not run on hardware
//
#include <hip/hip_runtime.h>
#include <hip/hip_bf16.h>
#include <math.h>

// ---------------------------------------------------------------------------
// Mamba block forward for MI455X (gfx1250): bf16 WMMA GEMMs (LDS-staged,
// async global->LDS double buffering) + fp32 VALU selective scan.
// ---------------------------------------------------------------------------

typedef __bf16 bf16_t;
typedef __attribute__((ext_vector_type(16))) __bf16 v16bf;
typedef __attribute__((ext_vector_type(8)))  __bf16 v8bf;
typedef __attribute__((ext_vector_type(8)))  float  v8f;
typedef __attribute__((ext_vector_type(4)))  int    v4i;

#define GLOBAL_AS __attribute__((address_space(1)))
#define LDS_AS    __attribute__((address_space(3)))

#if defined(__has_builtin)
#if __has_builtin(__builtin_amdgcn_global_load_async_to_lds_b128)
#define HAVE_ASYNC_LDS 1
#endif
#if __has_builtin(__builtin_amdgcn_s_wait_asynccnt)
#define HAVE_WAIT_ASYNC 1
#endif
#endif

// Problem constants
#define B_   2
#define L_   2048
#define DM_  2048
#define DI_  4096
#define DTR_ 128
#define DS_  16
#define DC_  4
#define M_   (B_ * L_)      // 4096 token rows
#define NPROJ_ (2 * DI_)    // 8192
#define NX_  256            // x_proj output padded 160 -> 256

// GEMM tiling
#define BM_T 128
#define BN_T 128
#define BK_T 32

// ---------------------------------------------------------------------------
// WMMA fragment loaders (bf16, 16x16x32), per CDNA5 ISA §7.12.2 layouts.
// ---------------------------------------------------------------------------

// A-matrix 16x32 bf16, row-major source (ld = row stride in elements).
// lanes 0-15: M=lane, K={0..7,16..23}; lanes 16-31: M=lane-16, K={8..15,24..31}
__device__ __forceinline__ v16bf load_a_frag(const bf16_t* A, int ld,
                                             int m0, int k0) {
  const int lane = threadIdx.x & 31;
  const int row  = m0 + (lane & 15);
  const int kb   = k0 + ((lane >> 4) << 3);
  const bf16_t* p = A + (size_t)row * (size_t)ld + kb;
  v8bf lo = *(const v8bf*)p;
  v8bf hi = *(const v8bf*)(p + 16);
  v16bf f;
#pragma unroll
  for (int i = 0; i < 8; ++i) { f[i] = lo[i]; f[8 + i] = hi[i]; }
  return f;
}

// B-matrix 32x16 bf16 from row-major (N x K) weight W: B(k,n) = W[n,k].
// lanes 0-15: N=lane, K=k0+{0..7,16..23}; lanes 16-31: N=lane-16, K=k0+{8..15,24..31}
__device__ __forceinline__ v16bf load_b_frag(const bf16_t* W, int ld,
                                             int n0, int k0) {
  const int lane = threadIdx.x & 31;
  const int col  = n0 + (lane & 15);
  const int kb   = k0 + ((lane >> 4) << 3);
  const bf16_t* p = W + (size_t)col * (size_t)ld + kb;
  v8bf lo = *(const v8bf*)p;
  v8bf hi = *(const v8bf*)(p + 16);
  v16bf f;
#pragma unroll
  for (int i = 0; i < 8; ++i) { f[i] = lo[i]; f[8 + i] = hi[i]; }
  return f;
}

// C/D 16x16 f32: lane 0-15 -> N=lane, M=v;  lane 16-31 -> N=lane-16, M=v+8
// mode 0: plain store.  mode 1: += bias[col], then softplus.
__device__ __forceinline__ void store_c_frag(float* __restrict__ C, int ld,
                                             int m0, int n0, v8f acc,
                                             int mode,
                                             const float* __restrict__ bias) {
  const int lane  = threadIdx.x & 31;
  const int col   = n0 + (lane & 15);
  const int rbase = m0 + ((lane >> 4) << 3);
  float bv = (mode == 1) ? bias[col] : 0.f;
#pragma unroll
  for (int v = 0; v < 8; ++v) {
    float val = acc[v];
    if (mode == 1) {
      val += bv;
      val = (val > 20.f) ? val : log1pf(__expf(val));  // softplus
    }
    C[(size_t)(rbase + v) * (size_t)ld + col] = val;
  }
}

// ---------------------------------------------------------------------------
// 16-byte global->LDS copy: async DMA path on gfx1250 (ASYNCcnt, bypasses
// VGPRs), register-staged fallback otherwise.
// ---------------------------------------------------------------------------
__device__ __forceinline__ void copy16_g2l(const bf16_t* src, bf16_t* dst) {
#if defined(HAVE_ASYNC_LDS)
  __builtin_amdgcn_global_load_async_to_lds_b128(
      (GLOBAL_AS v4i*)src, (LDS_AS v4i*)dst, 0, 0);
#else
  *(v8bf*)dst = *(const v8bf*)src;
#endif
}

template <int N>
__device__ __forceinline__ void wait_stage() {
#if defined(HAVE_ASYNC_LDS) && defined(HAVE_WAIT_ASYNC)
  __builtin_amdgcn_s_wait_asynccnt(N);
#endif
}

// Stage a 128x32 bf16 tile (row-major, ld elements) into LDS (8KB).
// 512 chunks of 16B; exactly 2 chunks per thread, branch-free.
__device__ __forceinline__ void stage_tile128(const bf16_t* g, int ld,
                                              bf16_t* l, int tid) {
  const int r = tid >> 2;
  const int p = (tid & 3) * 8;
  const bf16_t* gs = g + (size_t)r * (size_t)ld + p;
  bf16_t* ls = l + r * BK_T + p;
  copy16_g2l(gs, ls);                                       // rows 0..63
  copy16_g2l(gs + (size_t)64 * (size_t)ld, ls + 64 * BK_T); // rows 64..127
}

// One BK=32 k-step of the 32x64 wave tile: 2 A frags x 4 B frags = 8 WMMAs.
__device__ __forceinline__ void wmma_step(const bf16_t* la, const bf16_t* lb,
                                          int m0, int n0, v8f acc[2][4]) {
  v16bf a0 = load_a_frag(la, BK_T, m0,      0);
  v16bf a1 = load_a_frag(la, BK_T, m0 + 16, 0);
#pragma unroll
  for (int j = 0; j < 4; ++j) {
    v16bf bj = load_b_frag(lb, BK_T, n0 + j * 16, 0);
    acc[0][j] = __builtin_amdgcn_wmma_f32_16x16x32_bf16(
        false, a0, false, bj, (short)0, acc[0][j], false, false);
    acc[1][j] = __builtin_amdgcn_wmma_f32_16x16x32_bf16(
        false, a1, false, bj, (short)0, acc[1][j], false, false);
  }
}

// ---------------------------------------------------------------------------
// GEMM: C[M x N] (f32) = A[M x K] (bf16, row-major) * W[N x K]^T (bf16)
// 256 threads = 8 waves; block tile 128x128; wave tile 32x64 (2x4 frags).
// A/B tiles double-buffered in LDS; k-loop unrolled by 2 so buffer indices
// are compile-time constants (requires K % 64 == 0 -> nk even; true for all
// call sites: K in {2048, 4096, 128}).
// ---------------------------------------------------------------------------
__global__ __launch_bounds__(256) void mamba_gemm_bf16_wmma(
    const bf16_t* __restrict__ A, const bf16_t* __restrict__ W,
    float* __restrict__ C, int M, int N, int K, int mode,
    const float* __restrict__ bias) {
  __shared__ bf16_t lds_a[2][BM_T * BK_T];   // 2 x 8KB
  __shared__ bf16_t lds_b[2][BN_T * BK_T];   // 2 x 8KB

  const int tid  = threadIdx.x;
  const int wave = tid >> 5;
  const int wm = wave & 3;               // 4 waves along M
  const int wn = wave >> 2;              // 2 waves along N
  const int m0 = wm * 32;                // within block tile
  const int n0 = wn * 64;
  const int bm = blockIdx.y * BM_T;
  const int bn = blockIdx.x * BN_T;

  const bf16_t* Ab = A + (size_t)bm * (size_t)K;
  const bf16_t* Wb = W + (size_t)bn * (size_t)K;

  v8f acc[2][4];
#pragma unroll
  for (int i = 0; i < 2; ++i)
#pragma unroll
    for (int j = 0; j < 4; ++j) acc[i][j] = (v8f){};

  const int nk = K / BK_T;               // even by construction

  // Prologue: stage k-step 0 into buffer 0 (4 async ops per thread).
  stage_tile128(Ab, K, lds_a[0], tid);
  stage_tile128(Wb, K, lds_b[0], tid);

  for (int kk = 0; kk < nk; kk += 2) {
    // ---- phase 0: compute buffer 0, prefetch k-step kk+1 into buffer 1 ----
    if (kk + 1 < nk) {
      stage_tile128(Ab + (size_t)(kk + 1) * BK_T, K, lds_a[1], tid);
      stage_tile128(Wb + (size_t)(kk + 1) * BK_T, K, lds_b[1], tid);
      wait_stage<4>();   // oldest 4 (buffer 0) complete, newest 4 in flight
    } else {
      wait_stage<0>();
    }
    __syncthreads();
    wmma_step(lds_a[0], lds_b[0], m0, n0, acc);
    __syncthreads();

    // ---- phase 1: compute buffer 1, prefetch k-step kk+2 into buffer 0 ----
    if (kk + 1 < nk) {
      if (kk + 2 < nk) {
        stage_tile128(Ab + (size_t)(kk + 2) * BK_T, K, lds_a[0], tid);
        stage_tile128(Wb + (size_t)(kk + 2) * BK_T, K, lds_b[0], tid);
        wait_stage<4>();
      } else {
        wait_stage<0>();
      }
      __syncthreads();
      wmma_step(lds_a[1], lds_b[1], m0, n0, acc);
      __syncthreads();
    }
  }

#pragma unroll
  for (int i = 0; i < 2; ++i)
#pragma unroll
    for (int j = 0; j < 4; ++j)
      store_c_frag(C, N, bm + m0 + i * 16, bn + n0 + j * 16, acc[i][j],
                   mode, bias);
}

// ---------------------------------------------------------------------------
// f32 -> bf16 converters
// ---------------------------------------------------------------------------
__global__ void mamba_cvt_bf16(const float* __restrict__ src,
                               bf16_t* __restrict__ dst, int n) {
  int i = blockIdx.x * blockDim.x + threadIdx.x;
  if (i < n) dst[i] = (bf16_t)src[i];
}

// x_proj_w (160 x 4096) -> zero-padded (256 x 4096) bf16
__global__ void mamba_cvt_pad_xproj(const float* __restrict__ src,
                                    bf16_t* __restrict__ dst) {
  int i = blockIdx.x * blockDim.x + threadIdx.x;    // < 256*4096
  int row = i >> 12;
  int col = i & 4095;
  dst[i] = (row < (DTR_ + 2 * DS_)) ? (bf16_t)src[(size_t)row * DI_ + col]
                                    : (bf16_t)0.f;
}

// dt_lo: x_dbl[:, 0:128] (f32, ld 256) -> contiguous (4096 x 128) bf16
__global__ void mamba_cvt_dtlo(const float* __restrict__ xdbl,
                               bf16_t* __restrict__ dst) {
  int i = blockIdx.x * blockDim.x + threadIdx.x;    // < 4096*128
  int row = i >> 7;
  int col = i & 127;
  dst[i] = (bf16_t)xdbl[(size_t)row * NX_ + col];
}

// ---------------------------------------------------------------------------
// Depthwise causal conv (width 4) + bias + SiLU; also emits conv_state.
// One thread per (b, l, d). x = proj[:, 0:DI].
// ---------------------------------------------------------------------------
__global__ void mamba_conv_silu(const float* __restrict__ proj,
                                const float* __restrict__ conv_w,
                                const float* __restrict__ conv_b,
                                float* __restrict__ xc_f32,
                                bf16_t* __restrict__ xc_bf16,
                                float* __restrict__ conv_state) {
  const int idx = blockIdx.x * blockDim.x + threadIdx.x;  // < B*L*DI = 2^24
  const int d = idx & (DI_ - 1);
  const int l = (idx >> 12) & (L_ - 1);
  const int b = idx >> 23;
  const size_t rowbase = (size_t)(b * L_) * NPROJ_;

  float acc = conv_b[d];
#pragma unroll
  for (int j = 0; j < DC_; ++j) {
    int ls = l - (DC_ - 1) + j;
    if (ls >= 0)
      acc += conv_w[d * DC_ + j] * proj[rowbase + (size_t)ls * NPROJ_ + d];
  }
  float s = acc / (1.f + __expf(-acc));   // SiLU
  size_t o = (size_t)(b * L_ + l) * DI_ + d;
  xc_f32[o]  = s;
  xc_bf16[o] = (bf16_t)s;

  if (l >= L_ - DC_) {                    // raw x tail -> conv_state
    float xv = proj[rowbase + (size_t)l * NPROJ_ + d];
    conv_state[(size_t)b * DI_ * DC_ + (size_t)d * DC_ + (l - (L_ - DC_))] = xv;
  }
}

// ---------------------------------------------------------------------------
// Selective scan. One thread per (b, d) channel; 16-state recurrence in
// registers; B/C vectors staged per 16-step chunk into LDS.
// Fuses y = (y + x*D) * silu(z), emits bf16 A-operand for out_proj GEMM.
// ---------------------------------------------------------------------------
__global__ __launch_bounds__(256) void mamba_scan(
    const float* __restrict__ delta, const float* __restrict__ xconv,
    const float* __restrict__ xdbl,  const float* __restrict__ proj,
    const float* __restrict__ A_log, const float* __restrict__ Dvec,
    bf16_t* __restrict__ ybf, float* __restrict__ last_state) {
  const int b = blockIdx.x >> 4;                        // 2 batches
  const int d = ((blockIdx.x & 15) << 8) + threadIdx.x; // 16 blocks x 256 ch

  __shared__ float bc[16][32];   // [step][0:16]=B, [16:32]=C

  float negA[DS_];
#pragma unroll
  for (int s = 0; s < DS_; ++s) negA[s] = -__expf(A_log[d * DS_ + s]);
  const float Dv = Dvec[d];

  float h[DS_];
#pragma unroll
  for (int s = 0; s < DS_; ++s) h[s] = 0.f;

  for (int l0 = 0; l0 < L_; l0 += 16) {
    __syncthreads();
    for (int t = threadIdx.x; t < 512; t += 256) {
      int st = t >> 5, c = t & 31;
      int row = b * L_ + l0 + st;
      bc[st][c] = xdbl[(size_t)row * NX_ + DTR_ + c];
    }
    __syncthreads();

#pragma unroll 1
    for (int li = 0; li < 16; ++li) {
      const int row = b * L_ + l0 + li;
      const size_t o = (size_t)row * DI_ + d;
      const float dt  = delta[o];
      const float xv  = xconv[o];
      const float dtx = dt * xv;
      float y = 0.f;
#pragma unroll
      for (int s = 0; s < DS_; ++s) {
        float dA = __expf(dt * negA[s]);
        h[s] = dA * h[s] + dtx * bc[li][s];
        y += h[s] * bc[li][16 + s];
      }
      const float z  = proj[(size_t)row * NPROJ_ + DI_ + d];
      const float yv = (y + xv * Dv) * (z / (1.f + __expf(-z)));
      ybf[o] = (bf16_t)yv;
    }
  }

#pragma unroll
  for (int s = 0; s < DS_; ++s)
    last_state[(size_t)b * DI_ * DS_ + (size_t)d * DS_ + s] = h[s];
}

// ---------------------------------------------------------------------------
// Host-side orchestration
// ---------------------------------------------------------------------------
// Workspace layout (byte offsets, all 256B aligned)
#define OFF_HSBF   ((size_t)0)                         // 4096*2048 bf16
#define OFF_WINBF  (OFF_HSBF   + (size_t)16777216)     // 8192*2048 bf16
#define OFF_WXBF   (OFF_WINBF  + (size_t)33554432)     // 256*4096 bf16 (padded)
#define OFF_WDTBF  (OFF_WXBF   + (size_t)2097152)      // 4096*128 bf16
#define OFF_WOUTBF (OFF_WDTBF  + (size_t)1048576)      // 2048*4096 bf16
#define OFF_PROJ   (OFF_WOUTBF + (size_t)16777216)     // 4096*8192 f32
#define OFF_XCF    (OFF_PROJ   + (size_t)134217728)    // 4096*4096 f32
#define OFF_XCB    (OFF_XCF    + (size_t)67108864)     // 4096*4096 bf16
#define OFF_XDBL   (OFF_XCB    + (size_t)33554432)     // 4096*256 f32
#define OFF_DTLO   (OFF_XDBL   + (size_t)4194304)      // 4096*128 bf16
#define OFF_DELTA  (OFF_DTLO   + (size_t)1048576)      // 4096*4096 f32
#define OFF_YBF    (OFF_DELTA  + (size_t)67108864)     // 4096*4096 bf16

extern "C" void kernel_launch(void* const* d_in, const int* in_sizes, int n_in,
                              void* d_out, int out_size, void* d_ws,
                              size_t ws_size, hipStream_t stream) {
  const float* hidden    = (const float*)d_in[0];
  const float* in_proj_w = (const float*)d_in[1];
  const float* conv_w    = (const float*)d_in[2];
  const float* conv_b    = (const float*)d_in[3];
  const float* x_proj_w  = (const float*)d_in[4];
  const float* dt_proj_w = (const float*)d_in[5];
  const float* dt_proj_b = (const float*)d_in[6];
  const float* A_log     = (const float*)d_in[7];
  const float* Dvec      = (const float*)d_in[8];
  const float* out_proj_w= (const float*)d_in[9];

  char* ws = (char*)d_ws;
  bf16_t* hs_bf   = (bf16_t*)(ws + OFF_HSBF);
  bf16_t* win_bf  = (bf16_t*)(ws + OFF_WINBF);
  bf16_t* wx_bf   = (bf16_t*)(ws + OFF_WXBF);
  bf16_t* wdt_bf  = (bf16_t*)(ws + OFF_WDTBF);
  bf16_t* wout_bf = (bf16_t*)(ws + OFF_WOUTBF);
  float*  proj    = (float*)(ws + OFF_PROJ);
  float*  xcf     = (float*)(ws + OFF_XCF);
  bf16_t* xcb     = (bf16_t*)(ws + OFF_XCB);
  float*  xdbl    = (float*)(ws + OFF_XDBL);
  bf16_t* dtlo    = (bf16_t*)(ws + OFF_DTLO);
  float*  delta   = (float*)(ws + OFF_DELTA);
  bf16_t* ybf     = (bf16_t*)(ws + OFF_YBF);

  float* out        = (float*)d_out;                  // (2,2048,2048)
  float* conv_state = out + (size_t)B_ * L_ * DM_;    // (2,4096,4)
  float* last_state = conv_state + (size_t)B_ * DI_ * DC_;  // (2,4096,16)

  // 1) bf16 conversions of activations + weights
  mamba_cvt_bf16<<<(M_ * DM_) / 256, 256, 0, stream>>>(hidden, hs_bf, M_ * DM_);
  mamba_cvt_bf16<<<(NPROJ_ * DM_) / 256, 256, 0, stream>>>(in_proj_w, win_bf,
                                                           NPROJ_ * DM_);
  mamba_cvt_bf16<<<(DI_ * DTR_) / 256, 256, 0, stream>>>(dt_proj_w, wdt_bf,
                                                         DI_ * DTR_);
  mamba_cvt_bf16<<<(DM_ * DI_) / 256, 256, 0, stream>>>(out_proj_w, wout_bf,
                                                        DM_ * DI_);
  mamba_cvt_pad_xproj<<<(NX_ * DI_) / 256, 256, 0, stream>>>(x_proj_w, wx_bf);

  // 2) in_proj: proj[4096 x 8192] = hs * in_proj_w^T
  {
    dim3 g(NPROJ_ / BN_T, M_ / BM_T);
    mamba_gemm_bf16_wmma<<<g, 256, 0, stream>>>(hs_bf, win_bf, proj,
                                                M_, NPROJ_, DM_, 0, nullptr);
  }

  // 3) depthwise conv + SiLU (+ conv_state output)
  mamba_conv_silu<<<(B_ * L_ * DI_) / 256, 256, 0, stream>>>(
      proj, conv_w, conv_b, xcf, xcb, conv_state);

  // 4) x_proj: x_dbl[4096 x 256(pad)] = x_conv * x_proj_w^T
  {
    dim3 g(NX_ / BN_T, M_ / BM_T);
    mamba_gemm_bf16_wmma<<<g, 256, 0, stream>>>(xcb, wx_bf, xdbl,
                                                M_, NX_, DI_, 0, nullptr);
  }

  // 5) dt_lo -> bf16; delta = softplus(dt_lo * dt_proj_w^T + b)
  mamba_cvt_dtlo<<<(M_ * DTR_) / 256, 256, 0, stream>>>(xdbl, dtlo);
  {
    dim3 g(DI_ / BN_T, M_ / BM_T);
    mamba_gemm_bf16_wmma<<<g, 256, 0, stream>>>(dtlo, wdt_bf, delta,
                                                M_, DI_, DTR_, 1, dt_proj_b);
  }

  // 6) selective scan (fused D-skip + silu(z) gate), emits bf16 y
  mamba_scan<<<B_ * (DI_ / 256), 256, 0, stream>>>(
      delta, xcf, xdbl, proj, A_log, Dvec, ybf, last_state);

  // 7) out_proj: out[4096 x 2048] = y * out_proj_w^T
  {
    dim3 g(DM_ / BN_T, M_ / BM_T);
    mamba_gemm_bf16_wmma<<<g, 256, 0, stream>>>(ybf, wout_bf, out,
                                                M_, DM_, DI_, 0, nullptr);
  }
}
